// PointsDecoder_59450937311381
// MI455X (gfx1250) — matrix operand order repaired and checked by reading, not compiled, not run on hardware
//
#include <hip/hip_runtime.h>
#include <hip/hip_bf16.h>
#include <math.h>

// ---------------- problem constants ----------------
#define NB 4
#define MQ 32768
#define PP 1024
#define INDIM 32

typedef __attribute__((ext_vector_type(16))) _Float16 v16h;
typedef __attribute__((ext_vector_type(8)))  _Float16 v8h;
typedef __attribute__((ext_vector_type(8)))  float    v8f;

// activation panel width (halves) per wave-tile row
#define ACTW 192

// packed weight fragment bases (each fragment = 512 halves = 1KB)
#define FRAG_PLW1 0    // 71x64  -> Kc=3, Nn=4  (12)
#define FRAG_PLW2 12   // 64x32  -> Kc=2, Nn=2  (4)
#define FRAG_FW1  16   // 64x128 -> Kc=2, Nn=8  (16)
#define FRAG_FW2  32   // 128x128-> Kc=4, Nn=8  (32)
#define FRAG_FW3  64   // 128x128-> Kc=4, Nn=8  (32)
#define FRAG_RW1  96   // 155x64 -> Kc=5, Nn=4  (20)
#define FRAG_RW2  116  // 64x32  -> Kc=2, Nn=2  (4)
#define NFRAG     120

__device__ __forceinline__ float clamp1(float v) {
  return fminf(fmaxf(v, -0.999f), 0.999f);
}

// ---------------- weight repack: f32 row-major -> f16 B-fragments ----------------
// B-fragment layout (CDNA5 16-bit B 32x16): lane holds column N = lane&15 for
// K-half (lane>>4); element e -> K = kc*32 + (e&7) + 16*(e>>3) + 8*(lane>>4).
__global__ void repack_kernel(const float* __restrict__ w0, const float* __restrict__ w1,
                              const float* __restrict__ w2, const float* __restrict__ w3,
                              const float* __restrict__ w4, const float* __restrict__ w5,
                              const float* __restrict__ w6, _Float16* __restrict__ wpack) {
  int frag = blockIdx.x;
  int lane = threadIdx.x;
  const int baseArr[8] = {0, 12, 16, 32, 64, 96, 116, 120};
  const int KtArr[7]   = {71, 64, 64, 128, 128, 155, 64};
  const int NtArr[7]   = {64, 32, 128, 128, 128, 64, 32};
  const int KcArr[7]   = {3, 2, 2, 4, 4, 5, 2};
  int mi = 0;
  #pragma unroll
  for (int i = 0; i < 6; ++i) if (frag >= baseArr[i + 1]) mi = i + 1;
  const float* W = (mi == 0) ? w0 : (mi == 1) ? w1 : (mi == 2) ? w2 :
                   (mi == 3) ? w3 : (mi == 4) ? w4 : (mi == 5) ? w5 : w6;
  int local = frag - baseArr[mi];
  int Kc = KcArr[mi], Kt = KtArr[mi], Nt = NtArr[mi];
  int nn = local / Kc, kc = local % Kc;
  int ncol = nn * 16 + (lane & 15);
  _Float16* dst = wpack + (size_t)frag * 512 + lane * 16;
  #pragma unroll
  for (int e = 0; e < 16; ++e) {
    int k = kc * 32 + (e & 7) + ((e >> 3) << 4) + ((lane >> 4) << 3);
    float v = (k < Kt && ncol < Nt) ? W[(size_t)k * Nt + ncol] : 0.f;
    dst[e] = (_Float16)v;
  }
}

// ---------------- KNN: brute-force top-4 over 1024 points ----------------
__global__ __launch_bounds__(256) void knn_kernel(const float* __restrict__ coords,
                                                  const float* __restrict__ pts,
                                                  int* __restrict__ idxOut,
                                                  float* __restrict__ distOut) {
  __shared__ float sp[PP * 3];
  int n = blockIdx.y;
  const float* pbase = pts + (size_t)n * PP * 3;
  for (int i = threadIdx.x; i < PP * 3; i += 256) sp[i] = pbase[i];
  __syncthreads();
  int m = blockIdx.x * 256 + threadIdx.x;
  size_t q = (size_t)n * MQ + m;
  float cx = clamp1(coords[q * 3 + 0]);
  float cy = clamp1(coords[q * 3 + 1]);
  float cz = clamp1(coords[q * 3 + 2]);
  float b0 = 3.4e38f, b1 = 3.4e38f, b2 = 3.4e38f, b3 = 3.4e38f;
  int i0 = 0, i1 = 0, i2 = 0, i3 = 0;
  for (int p = 0; p < PP; ++p) {
    float dx = cx - sp[p * 3 + 0];
    float dy = cy - sp[p * 3 + 1];
    float dz = cz - sp[p * 3 + 2];
    float d = dx * dx + dy * dy + dz * dz;
    if (d < b3) {
      if (d < b2) {
        b3 = b2; i3 = i2;
        if (d < b1) {
          b2 = b1; i2 = i1;
          if (d < b0) { b1 = b0; i1 = i0; b0 = d; i0 = p; }
          else        { b1 = d;  i1 = p; }
        } else { b2 = d; i2 = p; }
      } else { b3 = d; i3 = p; }
    }
  }
  distOut[q * 4 + 0] = fmaxf(b0, 0.f);
  distOut[q * 4 + 1] = fmaxf(b1, 0.f);
  distOut[q * 4 + 2] = fmaxf(b2, 0.f);
  distOut[q * 4 + 3] = fmaxf(b3, 0.f);
  idxOut[q * 4 + 0] = i0; idxOut[q * 4 + 1] = i1;
  idxOut[q * 4 + 2] = i2; idxOut[q * 4 + 3] = i3;
}

// ---------------- tri-plane bilinear sampling ----------------
__global__ __launch_bounds__(256) void tex_kernel(const float* __restrict__ coords,
                                                  const float* __restrict__ planes,
                                                  float* __restrict__ texbuf) {
  int gid = blockIdx.x * 256 + threadIdx.x;
  int c = gid & 31;
  size_t q = (size_t)(gid >> 5);
  int n = (int)(q / MQ);
  float x = clamp1(coords[q * 3 + 0]);
  float y = clamp1(coords[q * 3 + 1]);
  float z = clamp1(coords[q * 3 + 2]);
  float acc = 0.f;
  #pragma unroll
  for (int pl = 0; pl < 3; ++pl) {
    float u = (pl == 2) ? z : x;                 // proj: (x,y), (x,z), (z,x)
    float v = (pl == 0) ? y : ((pl == 1) ? z : x);
    float fx = (u + 1.f) * 128.f - 0.5f;
    float fy = (v + 1.f) * 128.f - 0.5f;
    float x0f = floorf(fx), y0f = floorf(fy);
    float wx = fx - x0f, wy = fy - y0f;
    int x0 = (int)x0f, y0 = (int)y0f;
    int x0i = x0 < 0 ? 0 : (x0 > 255 ? 255 : x0);
    int x1i = (x0 + 1) < 0 ? 0 : ((x0 + 1) > 255 ? 255 : x0 + 1);
    int y0i = y0 < 0 ? 0 : (y0 > 255 ? 255 : y0);
    int y1i = (y0 + 1) < 0 ? 0 : ((y0 + 1) > 255 ? 255 : y0 + 1);
    const float* fp = planes + (((size_t)n * 3 + pl) * 32 + c) * 65536;
    float v00 = fp[y0i * 256 + x0i];
    float v01 = fp[y0i * 256 + x1i];
    float v10 = fp[y1i * 256 + x0i];
    float v11 = fp[y1i * 256 + x1i];
    acc += v00 * (1 - wx) * (1 - wy) + v01 * wx * (1 - wy) +
           v10 * (1 - wx) * wy + v11 * wx * wy;
  }
  texbuf[q * 32 + c] = acc * (1.f / 3.f);
}

// ---------------- WMMA fragment helpers ----------------
// A-fragment (CDNA5 16-bit A 16x32): row = lane&15;
// element e -> K = koff + kc*32 + e + 8*((e>>3) + (lane>>4))
__device__ __forceinline__ v16h load_a(const _Float16* actW, int koff, int kc, int lane) {
  const _Float16* p = actW + (size_t)(lane & 15) * ACTW + koff + kc * 32 + ((lane >> 4) << 3);
  v8h lo = *(const v8h*)p;
  v8h hi = *(const v8h*)(p + 16);
  v16h a;
  #pragma unroll
  for (int e = 0; e < 8; ++e) { a[e] = lo[e]; a[e + 8] = hi[e]; }
  return a;
}

__device__ __forceinline__ v16h load_b(const _Float16* wpack, int frag, int lane) {
  return *(const v16h*)(wpack + (size_t)frag * 512 + lane * 16);
}

__device__ __forceinline__ v8f wmma_row(const _Float16* actW, int koff, int kcCnt,
                                        const _Float16* wpack, int fragBase, int Kc,
                                        int nt, int lane, float bias) {
  v8f c;
  #pragma unroll
  for (int r = 0; r < 8; ++r) c[r] = bias;
  for (int kc = 0; kc < kcCnt; ++kc) {
    v16h a = load_a(actW, koff, kc, lane);
    v16h b = load_b(wpack, fragBase + nt * Kc + kc, lane);
    c = __builtin_amdgcn_wmma_f32_16x16x32_f16(false, a, false, b, (short)0, c, false, false);
  }
  return c;
}

// C/D layout: element r -> row = r + 8*(lane>>4), col = lane&15
__device__ __forceinline__ void store_c(_Float16* actW, int colOff, int nt, int lane,
                                        v8f c, bool relu) {
  int col = colOff + nt * 16 + (lane & 15);
  int r0 = (lane >> 4) * 8;
  #pragma unroll
  for (int r = 0; r < 8; ++r) {
    float v = c[r];
    if (relu) v = fmaxf(v, 0.f);
    actW[(size_t)(r0 + r) * ACTW + col] = (_Float16)v;
  }
}

// ---------------- fused MLP chain: 4 waves/block, 16 queries per wave ----------------
__global__ __launch_bounds__(128) void mlp_kernel(
    const float* __restrict__ coords, const float* __restrict__ dirs,
    const float* __restrict__ ppos, const float* __restrict__ pfeat,
    const float* __restrict__ texbuf, const int* __restrict__ idxbuf,
    const float* __restrict__ dist, const _Float16* __restrict__ wpack,
    const float* __restrict__ plb1, const float* __restrict__ plb2,
    const float* __restrict__ fb1, const float* __restrict__ fb2,
    const float* __restrict__ fb3, const float* __restrict__ dwp,
    const float* __restrict__ dbp, const float* __restrict__ rb1,
    const float* __restrict__ rb2, float* __restrict__ dens,
    float* __restrict__ rgb) {
  __shared__ __align__(16) _Float16 actAll[4][16 * ACTW];
  __shared__ float scoord[4][16][3];
  __shared__ float sdir[4][16][3];
  __shared__ float swq[4][16][4];
  __shared__ int   sidx[4][16][4];

  int wave = threadIdx.x >> 5;
  int lane = threadIdx.x & 31;
  int qbase = blockIdx.x * 64 + wave * 16;
  int n = qbase / MQ;
  _Float16* actW = &actAll[wave][0];

  // ---- per-tile setup (one query per lane 0..15) ----
  if (lane < 16) {
    int row = lane;
    size_t q = (size_t)qbase + row;
    scoord[wave][row][0] = clamp1(coords[q * 3 + 0]);
    scoord[wave][row][1] = clamp1(coords[q * 3 + 1]);
    scoord[wave][row][2] = clamp1(coords[q * 3 + 2]);
    float d0 = dirs[q * 3 + 0], d1 = dirs[q * 3 + 1], d2 = dirs[q * 3 + 2];
    float nrm = fmaxf(sqrtf(d0 * d0 + d1 * d1 + d2 * d2), 1e-12f);
    sdir[wave][row][0] = d0 / nrm;
    sdir[wave][row][1] = d1 / nrm;
    sdir[wave][row][2] = d2 / nrm;
    float wv[4]; float wsum = 0.f;
    #pragma unroll
    for (int j = 0; j < 4; ++j) {
      float dd = dist[q * 4 + j];
      wv[j] = 1.f / dd;
      wsum += wv[j];
      sidx[wave][row][j] = idxbuf[q * 4 + j];
    }
    #pragma unroll
    for (int j = 0; j < 4; ++j) swq[wave][row][j] = wv[j] / wsum;
  }
  __syncthreads();

  int row = lane & 15;
  int half = lane >> 4;
  int r0 = half * 8;

  // ---- per-neighbor point MLP (71->64->32), accumulate inv-dist weighted ----
  v8f ptsC[2];
  #pragma unroll
  for (int nt = 0; nt < 2; ++nt)
    #pragma unroll
    for (int r = 0; r < 8; ++r) ptsC[nt][r] = 0.f;

  for (int j = 0; j < 4; ++j) {
    // fill A panel cols 0..95: feats(32) + harmonic-embed(39) + zero pad
    {
      int pid = sidx[wave][row][j];
      const float* pp = ppos + ((size_t)n * PP + pid) * 3;
      float rel0 = scoord[wave][row][0] - pp[0];
      float rel1 = scoord[wave][row][1] - pp[1];
      float rel2 = scoord[wave][row][2] - pp[2];
      const float* pf = pfeat + ((size_t)n * PP + pid) * INDIM;
      _Float16* arow = actW + (size_t)row * ACTW;
      int k0 = half * 48;
      for (int k = k0; k < k0 + 48; ++k) {
        float v;
        if (k < 32) v = pf[k];
        else if (k < 71) {
          int e = k - 32;
          if (e < 18) {
            int d = e / 6, fr = e % 6;
            float x = (d == 0) ? rel0 : ((d == 1) ? rel1 : rel2);
            v = __sinf(x * (float)(1 << fr));
          } else if (e < 36) {
            int e2 = e - 18; int d = e2 / 6, fr = e2 % 6;
            float x = (d == 0) ? rel0 : ((d == 1) ? rel1 : rel2);
            v = __cosf(x * (float)(1 << fr));
          } else v = (e == 36) ? rel0 : ((e == 37) ? rel1 : rel2);
        } else v = 0.f;
        arow[k] = (_Float16)v;
      }
    }
    __syncthreads();
    // pl layer 1: 16x96 @ 96x64, relu -> cols 96..159
    #pragma unroll
    for (int nt = 0; nt < 4; ++nt) {
      v8f h1 = wmma_row(actW, 0, 3, wpack, FRAG_PLW1, 3, nt, lane,
                        plb1[nt * 16 + (lane & 15)]);
      store_c(actW, 96, nt, lane, h1, true);
    }
    __syncthreads();
    // pl layer 2: 16x64 @ 64x32, accumulate weighted
    #pragma unroll
    for (int nt = 0; nt < 2; ++nt) {
      v8f h2 = wmma_row(actW, 96, 2, wpack, FRAG_PLW2, 2, nt, lane,
                        plb2[nt * 16 + (lane & 15)]);
      #pragma unroll
      for (int r = 0; r < 8; ++r) ptsC[nt][r] += h2[r] * swq[wave][r0 + r][j];
    }
    __syncthreads();
  }

  // ---- f-input: cols 0..31 tex, 32..63 pts_feat ----
  {
    size_t q = (size_t)qbase + row;
    const float* tq = texbuf + q * 32 + half * 16;
    _Float16* arow = actW + (size_t)row * ACTW + half * 16;
    for (int c = 0; c < 16; ++c) arow[c] = (_Float16)tq[c];
  }
  #pragma unroll
  for (int nt = 0; nt < 2; ++nt) {
    int col = 32 + nt * 16 + (lane & 15);
    #pragma unroll
    for (int r = 0; r < 8; ++r)
      actW[(size_t)(r0 + r) * ACTW + col] = (_Float16)ptsC[nt][r];
  }
  __syncthreads();

  // f1: 16x64 @ 64x128, relu -> cols 64..191
  #pragma unroll
  for (int nt = 0; nt < 8; ++nt) {
    v8f h = wmma_row(actW, 0, 2, wpack, FRAG_FW1, 2, nt, lane,
                     fb1[nt * 16 + (lane & 15)]);
    store_c(actW, 64, nt, lane, h, true);
  }
  __syncthreads();

  // f2: 16x128 @ 128x128, relu -> cols 0..127
  v8f f2c[8];
  #pragma unroll
  for (int nt = 0; nt < 8; ++nt)
    f2c[nt] = wmma_row(actW, 64, 4, wpack, FRAG_FW2, 4, nt, lane,
                       fb2[nt * 16 + (lane & 15)]);
  __syncthreads();
  #pragma unroll
  for (int nt = 0; nt < 8; ++nt) store_c(actW, 0, nt, lane, f2c[nt], true);
  __syncthreads();

  // f3: 16x128 @ 128x128 (no relu), keep in registers
  v8f fC[8];
  #pragma unroll
  for (int nt = 0; nt < 8; ++nt)
    fC[nt] = wmma_row(actW, 0, 4, wpack, FRAG_FW3, 4, nt, lane,
                      fb3[nt * 16 + (lane & 15)]);
  __syncthreads();

  // stage f (cols 0..127) + rays_emb (cols 128..159) for the rgb head
  #pragma unroll
  for (int nt = 0; nt < 8; ++nt) store_c(actW, 0, nt, lane, fC[nt], false);
  {
    float dx = sdir[wave][row][0], dy = sdir[wave][row][1], dz = sdir[wave][row][2];
    _Float16* arow = actW + (size_t)row * ACTW;
    int k0 = 128 + half * 16;
    for (int k = k0; k < k0 + 16; ++k) {
      int e = k - 128; float v;
      if (e < 12) {
        int d = e / 4, fr = e % 4;
        float x = (d == 0) ? dx : ((d == 1) ? dy : dz);
        v = __sinf(x * (float)(1 << fr));
      } else if (e < 24) {
        int e2 = e - 12; int d = e2 / 4, fr = e2 % 4;
        float x = (d == 0) ? dx : ((d == 1) ? dy : dz);
        v = __cosf(x * (float)(1 << fr));
      } else if (e < 27) v = (e == 24) ? dx : ((e == 25) ? dy : dz);
      else v = 0.f;
      arow[k] = (_Float16)v;
    }
  }

  // density head from fC registers (f32 precision): reduce over 128 cols
  {
    float ds[8];
    #pragma unroll
    for (int r = 0; r < 8; ++r) ds[r] = 0.f;
    #pragma unroll
    for (int nt = 0; nt < 8; ++nt) {
      float w = dwp[nt * 16 + (lane & 15)];
      #pragma unroll
      for (int r = 0; r < 8; ++r) ds[r] += fC[nt][r] * w;
    }
    for (int msk = 1; msk < 16; msk <<= 1) {
      #pragma unroll
      for (int r = 0; r < 8; ++r) ds[r] += __shfl_xor(ds[r], msk, 32);
    }
    if ((lane & 15) == 0) {
      float db = dbp[0];
      #pragma unroll
      for (int r = 0; r < 8; ++r) {
        float t = 10.f * (ds[r] + db);
        float sp = (t > 20.f) ? t : log1pf(__expf(t));
        float raw = sp * 0.1f;
        dens[(size_t)qbase + r0 + r] = 1.f - __expf(-raw);
      }
    }
  }
  __syncthreads();

  // r1: 16x160 @ 160x64, relu -> cols 128..191
  v8f r1c[4];
  #pragma unroll
  for (int nt = 0; nt < 4; ++nt)
    r1c[nt] = wmma_row(actW, 0, 5, wpack, FRAG_RW1, 5, nt, lane,
                       rb1[nt * 16 + (lane & 15)]);
  __syncthreads();
  #pragma unroll
  for (int nt = 0; nt < 4; ++nt) store_c(actW, 128, nt, lane, r1c[nt], true);
  __syncthreads();

  // r2: 16x64 @ 64x32 -> rgb output (sigmoid remap on first 3 cols)
  #pragma unroll
  for (int nt = 0; nt < 2; ++nt) {
    v8f c = wmma_row(actW, 128, 2, wpack, FRAG_RW2, 2, nt, lane,
                     rb2[nt * 16 + (lane & 15)]);
    int col = nt * 16 + (lane & 15);
    #pragma unroll
    for (int r = 0; r < 8; ++r) {
      float v = c[r];
      if (col < 3) {
        v = 1.f / (1.f + __expf(-v));
        v = v * 1.002f - 0.001f;
      }
      rgb[((size_t)qbase + r0 + r) * 32 + col] = v;
    }
  }
}

// ---------------- launcher ----------------
extern "C" void kernel_launch(void* const* d_in, const int* in_sizes, int n_in,
                              void* d_out, int out_size, void* d_ws, size_t ws_size,
                              hipStream_t stream) {
  const float* coords = (const float*)d_in[0];
  const float* dirs   = (const float*)d_in[1];
  const float* ppos   = (const float*)d_in[2];
  const float* pfeat  = (const float*)d_in[3];
  const float* planes = (const float*)d_in[4];
  const float* pl_w1  = (const float*)d_in[5];
  const float* pl_b1  = (const float*)d_in[6];
  const float* pl_w2  = (const float*)d_in[7];
  const float* pl_b2  = (const float*)d_in[8];
  const float* f_w1   = (const float*)d_in[9];
  const float* f_b1   = (const float*)d_in[10];
  const float* f_w2   = (const float*)d_in[11];
  const float* f_b2   = (const float*)d_in[12];
  const float* f_w3   = (const float*)d_in[13];
  const float* f_b3   = (const float*)d_in[14];
  const float* d_w    = (const float*)d_in[15];
  const float* d_b    = (const float*)d_in[16];
  const float* r_w1   = (const float*)d_in[17];
  const float* r_b1   = (const float*)d_in[18];
  const float* r_w2   = (const float*)d_in[19];
  const float* r_b2   = (const float*)d_in[20];

  char* ws = (char*)d_ws;
  _Float16* wpack = (_Float16*)ws;                       // 120 KB
  int* idxbuf = (int*)(ws + 122880);                     // 2 MB
  float* texbuf = (float*)(ws + 122880 + 2097152);       // 16 MB

  float* dens = (float*)d_out;                           // N*M
  float* rgb = dens + (size_t)NB * MQ;                   // N*M*32
  float* distOut = rgb + (size_t)NB * MQ * 32;           // N*M*4

  repack_kernel<<<NFRAG, 32, 0, stream>>>(pl_w1, pl_w2, f_w1, f_w2, f_w3, r_w1, r_w2, wpack);
  knn_kernel<<<dim3(MQ / 256, NB), 256, 0, stream>>>(coords, ppos, idxbuf, distOut);
  tex_kernel<<<(NB * MQ * 32) / 256, 256, 0, stream>>>(coords, planes, texbuf);
  mlp_kernel<<<(NB * MQ) / 64, 128, 0, stream>>>(
      coords, dirs, ppos, pfeat, texbuf, idxbuf, distOut, wpack,
      pl_b1, pl_b2, f_b1, f_b2, f_b3, d_w, d_b, r_b1, r_b2, dens, rgb);
}
